// AttnModule_18141941858958
// MI455X (gfx1250) — compile-verified, matrix operand order checked by reading, not run on hardware
//
#include <hip/hip_runtime.h>
#include <math.h>

// ---------------------------------------------------------------------------
// Graph attention for MI455X (gfx1250): bf16 WMMA everywhere, flash-style
// fused attention with TDM (tensor_load_to_lds) double-buffered K/V staging.
//   B=16, N=512, DIN=128, H=8, DH=64, DO=64, DLIN=128
// ---------------------------------------------------------------------------

#define B_    16
#define N_    512
#define DIN_  128
#define H_    8
#define DH_   64
#define DO_   64
#define DLIN_ 128
#define NEG_  (-1e9f)

typedef __attribute__((ext_vector_type(16))) __bf16 v16bf;
typedef __attribute__((ext_vector_type(8)))  __bf16 v8bf;
typedef __attribute__((ext_vector_type(8)))  float  v8f;
typedef __attribute__((ext_vector_type(4)))  unsigned int v4u;
typedef __attribute__((ext_vector_type(8)))  int v8i_;
typedef __attribute__((ext_vector_type(4)))  int v4i_;

#if __has_builtin(__builtin_amdgcn_tensor_load_to_lds) && \
    __has_builtin(__builtin_amdgcn_s_wait_tensorcnt)
#define HAVE_TDM 1
#else
#define HAVE_TDM 0
#endif

// Load a 16x32 bf16 fragment in the CDNA5 WMMA A/B register layout.
// base points at a [16 rows x K] tile stored row-major with leading dim `ld`
// (for B fragments, "rows" are the N dimension and memory holds B^T).
// Lane l: row = l&15, kbase = (l>>4)*8; elems 0..7 -> K=kbase..kbase+7,
// elems 8..15 -> K=kbase+16..kbase+23  (two contiguous 16B loads per lane).
static __device__ __forceinline__ v16bf load_frag(const __bf16* base, int ld) {
  const int l  = threadIdx.x & 31;
  const int rn = l & 15;
  const int kb = (l >> 4) << 3;
  const __bf16* p = base + rn * ld + kb;
  v8bf lo = *(const v8bf*)(p);
  v8bf hi = *(const v8bf*)(p + 16);
  v16bf a;
#pragma unroll
  for (int i = 0; i < 8; ++i) { a[i] = lo[i]; a[i + 8] = hi[i]; }
  return a;
}

static __device__ __forceinline__ v8f wmma_bf(v16bf a, v16bf b, v8f c) {
  // D = A(16x32) * B(32x16) + C(16x16), f32 accumulate
  return __builtin_amdgcn_wmma_f32_16x16x32_bf16(
      false, a, false, b, (short)0, c, false, false);
}

#if HAVE_TDM
// Issue one TDM 2D tile load (bf16 elements) from global to LDS.
// tile: tile_d0 contiguous elems per row, tile_d1 rows, row stride = stride0
// elems. Descriptor built from uniform (SGPR) values; EXEC is ignored by TDM.
static __device__ __forceinline__ void tdm_load_2d(unsigned int lds_off,
                                                   const void* gsrc,
                                                   unsigned int tile_d0,
                                                   unsigned int tile_d1,
                                                   unsigned int tensor_d0,
                                                   unsigned int tensor_d1,
                                                   unsigned long long stride0) {
  unsigned long long ga = (unsigned long long)gsrc;
  v4u g0;
  g0[0] = 1u;                                   // count=1, user descriptor
  g0[1] = lds_off;                              // LDS byte address
  g0[2] = (unsigned int)ga;                     // global_addr[31:0]
  g0[3] = (unsigned int)(ga >> 32) | (2u << 30);// global_addr[56:32] | type=2
  v8i_ g1;
  g1[0] = (int)(1u << 16);                      // data_size=1 (2 bytes)
  g1[1] = (int)(tensor_d0 << 16);               // [63:48] tensor_dim0 lo16
  g1[2] = (int)((tensor_d0 >> 16) | (tensor_d1 << 16));
  g1[3] = (int)((tensor_d1 >> 16) | (tile_d0 << 16));
  g1[4] = (int)tile_d1;                         // tile_dim1; tile_dim2=0
  g1[5] = (int)(unsigned int)(stride0 & 0xffffffffu);
  g1[6] = (int)(unsigned int)((stride0 >> 32) & 0xffffu); // stride1=0
  g1[7] = 0;
  v4i_ z4 = {0, 0, 0, 0};
#if __clang_major__ >= 23
  v8i_ z8 = {0, 0, 0, 0, 0, 0, 0, 0};
  __builtin_amdgcn_tensor_load_to_lds(g0, g1, z4, z4, z8, 0);
#else
  __builtin_amdgcn_tensor_load_to_lds(g0, g1, z4, z4, 0);
#endif
}
#endif  // HAVE_TDM

// -------------------------------- prep -------------------------------------

__global__ void cast_bf16_kernel(const float* __restrict__ s,
                                 __bf16* __restrict__ d, int n) {
  int i = blockIdx.x * blockDim.x + threadIdx.x;
  if (i < n) d[i] = (__bf16)s[i];
}

// s: [batch, rows, cols] f32  ->  d: [batch, cols, rows] bf16 (transposed)
__global__ void transpose_bf16_kernel(const float* __restrict__ s,
                                      __bf16* __restrict__ d,
                                      int batch, int rows, int cols) {
  int i = blockIdx.x * blockDim.x + threadIdx.x;
  int per = rows * cols;
  if (i >= batch * per) return;
  int b = i / per;
  int rem = i - b * per;
  int c = rem / rows;
  int r = rem - c * rows;
  d[i] = (__bf16)s[(b * rows + r) * cols + c];
}

// ------------------------------ QKV projection -----------------------------
// grid (N/16, B*H, 3), block 128 (4 waves; wave = 16-col tile of DH/DO)
__global__ __launch_bounds__(128)
void qkv_kernel(const __bf16* __restrict__ Xh,
                const __bf16* __restrict__ Wqt, const __bf16* __restrict__ Wkt,
                const __bf16* __restrict__ Wvt,
                const float* __restrict__ bq, const float* __restrict__ bk,
                const float* __restrict__ bv,
                __bf16* __restrict__ Q, __bf16* __restrict__ K,
                __bf16* __restrict__ Vt) {
  const int ct = threadIdx.x >> 5;             // column tile 0..3
  const int l  = threadIdx.x & 31;
  const int ln = l & 15, lg = l >> 4;
  const int qt = blockIdx.x;                   // 16-row tile 0..31
  const int bh = blockIdx.y;                   // b*H + h
  const int b  = bh >> 3, h = bh & 7;
  const int mat = blockIdx.z;                  // 0=Q 1=K 2=V

  const __bf16* W    = (mat == 0 ? Wqt : mat == 1 ? Wkt : Wvt) + h * (DH_ * DIN_);
  const float*  bias = (mat == 0 ? bq  : mat == 1 ? bk  : bv ) + h * DH_;

  float bb = bias[ct * 16 + ln];
  v8f c;
#pragma unroll
  for (int r = 0; r < 8; ++r) c[r] = bb;

  const __bf16* Xb = Xh + (size_t)(b * N_ + qt * 16) * DIN_;
#pragma unroll
  for (int k0 = 0; k0 < DIN_; k0 += 32) {
    v16bf a  = load_frag(Xb + k0, DIN_);
    v16bf wb = load_frag(W + ct * 16 * DIN_ + k0, DIN_);
    c = wmma_bf(a, wb, c);
  }

  if (mat < 2) {
    __bf16* out = (mat == 0 ? Q : K) + (size_t)bh * N_ * DH_;
#pragma unroll
    for (int r = 0; r < 8; ++r) {
      int row = qt * 16 + r + 8 * lg;
      out[row * DH_ + ct * 16 + ln] = (__bf16)c[r];
    }
  } else {
    // store V transposed: Vt[bh, e, key]; C/D rows are consecutive keys ->
    // one packed contiguous 8x bf16 store per lane.
    v8bf p;
#pragma unroll
    for (int r = 0; r < 8; ++r) p[r] = (__bf16)c[r];
    *(v8bf*)(Vt + ((size_t)(bh * DO_ + ct * 16 + ln)) * N_ + qt * 16 + 8 * lg) = p;
  }
}

// ----------------------------- fused attention -----------------------------
// grid (N/16, B*H), block 32: one wave owns one 16-query tile, streams keys
// 32 at a time with masked online softmax.  K/V blocks are staged into LDS
// double-buffers by the Tensor Data Mover (TENSORcnt-tracked), overlapping
// the next block's DMA with the current block's 8 WMMAs.
__global__ __launch_bounds__(32)
void attn_kernel(const __bf16* __restrict__ Q, const __bf16* __restrict__ K,
                 const __bf16* __restrict__ Vt,
                 const int* __restrict__ adj, const int* __restrict__ masks,
                 __bf16* __restrict__ AO) {
  __shared__ __align__(16) __bf16 Pbuf[16 * 32];
#if HAVE_TDM
  __shared__ __align__(16) __bf16 Kbuf[2][32 * DH_];   // 2 x 4KB
  __shared__ __align__(16) __bf16 Vbuf[2][DO_ * 32];   // 2 x 4KB
#endif

  const int qt = blockIdx.x;
  const int bh = blockIdx.y;
  const int b  = bh >> 3, h = bh & 7;
  const int l  = threadIdx.x & 31;
  const int ln = l & 15, lg = l >> 4;

  const __bf16* Qb  = Q  + (size_t)(bh * N_ + qt * 16) * DH_;
  const __bf16* Kb  = K  + (size_t)bh * N_ * DH_;
  const __bf16* Vb  = Vt + (size_t)bh * DO_ * N_;
  const int* adjb   = adj + (size_t)b * N_ * N_;
  const int* mb     = masks + b * N_;

  v16bf qa0 = load_frag(Qb, DH_);        // e = 0..31
  v16bf qa1 = load_frag(Qb + 32, DH_);   // e = 32..63

  float m[8], lsum[8];
  v8f O[4];
#pragma unroll
  for (int r = 0; r < 8; ++r) { m[r] = -__builtin_inff(); lsum[r] = 0.f; }
#pragma unroll
  for (int et = 0; et < 4; ++et)
#pragma unroll
    for (int r = 0; r < 8; ++r) O[et][r] = 0.f;

#if HAVE_TDM
  // Prologue: DMA block 0 (K: 32 rows x 64 contiguous; V^T: 64 rows x 32,
  // row stride N_) into LDS buffer 0.  LDS offset = low 32 bits of the flat
  // shared-memory address (LDS aperture maps addr[31:0] -> LDS offset).
  tdm_load_2d((unsigned int)(size_t)&Kbuf[0][0], Kb, DH_, 32, DH_, N_, DH_);
  tdm_load_2d((unsigned int)(size_t)&Vbuf[0][0], Vb, 32, DO_, 32, DO_, N_);
  asm volatile("" ::: "memory");
#endif

  for (int ks = 0; ks < N_ / 32; ++ks) {
    const int key0 = ks * 32;
#if HAVE_TDM
    const int buf = ks & 1;
    if (ks + 1 < N_ / 32) {
      // DMA next block into the other buffer while we compute on this one.
      tdm_load_2d((unsigned int)(size_t)&Kbuf[buf ^ 1][0],
                  Kb + (size_t)(key0 + 32) * DH_, DH_, 32, DH_, N_, DH_);
      tdm_load_2d((unsigned int)(size_t)&Vbuf[buf ^ 1][0],
                  Vb + key0 + 32, 32, DO_, 32, DO_, N_);
      __builtin_amdgcn_s_wait_tensorcnt(2);   // this block's pair is done
    } else {
      __builtin_amdgcn_s_wait_tensorcnt(0);
    }
    asm volatile("" ::: "memory");
    const __bf16* Kf  = &Kbuf[buf][0];  const int koff = 0;
    const __bf16* Vf  = &Vbuf[buf][0];  const int ldv  = 32;
#else
    const __bf16* Kf  = Kb + (size_t)key0 * DH_;  const int koff = key0;
    const __bf16* Vf  = Vb;                        const int ldv  = N_;
#endif
    // prefetch next adjacency block for this lane's columns
    if (ks + 1 < N_ / 32)
      __builtin_prefetch(adjb + (size_t)(qt * 16 + 8 * lg) * N_ + key0 + 32 + ln,
                         0, 1);

    // ---- S = Q K^T for 32 keys (two 16x16 tiles) ----
    v8f S0, S1;
#pragma unroll
    for (int r = 0; r < 8; ++r) { S0[r] = 0.f; S1[r] = 0.f; }
    v16bf b00 = load_frag(Kf, DH_);
    v16bf b01 = load_frag(Kf + 32, DH_);
    v16bf b10 = load_frag(Kf + 16 * DH_, DH_);
    v16bf b11 = load_frag(Kf + 16 * DH_ + 32, DH_);
    S0 = wmma_bf(qa0, b00, S0); S0 = wmma_bf(qa1, b01, S0);
    S1 = wmma_bf(qa0, b10, S1); S1 = wmma_bf(qa1, b11, S1);

    // ---- scale + adjacency/node masking (C/D layout) ----
    const int kc0 = key0 + ln, kc1 = key0 + 16 + ln;
    const int mk0 = mb[kc0], mk1 = mb[kc1];
    float s0[8], s1[8];
#pragma unroll
    for (int r = 0; r < 8; ++r) {
      int qrow = qt * 16 + r + 8 * lg;
      int a0 = adjb[(size_t)qrow * N_ + kc0];
      int a1 = adjb[(size_t)qrow * N_ + kc1];
      s0[r] = (a0 && mk0) ? S0[r] * 0.125f : NEG_;   // 1/sqrt(64)
      s1[r] = (a1 && mk1) ? S1[r] * 0.125f : NEG_;
    }

    // ---- online softmax (rows live across 16-lane halves) ----
    float alpha[8];
#pragma unroll
    for (int r = 0; r < 8; ++r) {
      float bm = fmaxf(s0[r], s1[r]);
#pragma unroll
      for (int msk = 1; msk < 16; msk <<= 1)
        bm = fmaxf(bm, __shfl_xor(bm, msk, 32));
      float mn = fmaxf(m[r], bm);
      alpha[r] = __expf(m[r] - mn);
      float p0 = __expf(s0[r] - mn);
      float p1 = __expf(s1[r] - mn);
      float rs = p0 + p1;
#pragma unroll
      for (int msk = 1; msk < 16; msk <<= 1)
        rs += __shfl_xor(rs, msk, 32);
      lsum[r] = lsum[r] * alpha[r] + rs;
      m[r] = mn;
      int row = r + 8 * lg;
      Pbuf[row * 32 + ln]      = (__bf16)p0;   // C/D -> row-major P in LDS
      Pbuf[row * 32 + 16 + ln] = (__bf16)p1;
    }
#pragma unroll
    for (int et = 0; et < 4; ++et)
#pragma unroll
      for (int r = 0; r < 8; ++r) O[et][r] *= alpha[r];

    __syncthreads();  // order LDS stores vs. transposed A-fragment loads
    v16bf pa = load_frag(Pbuf, 32);            // P as A-matrix (16x32)
#pragma unroll
    for (int et = 0; et < 4; ++et) {
      v16bf vb = load_frag(Vf + (size_t)(et * 16) * ldv + koff, ldv);
      O[et] = wmma_bf(pa, vb, O[et]);
    }
    __syncthreads();
  }

  // ---- normalize and write concat-head output (bf16) ----
#pragma unroll
  for (int et = 0; et < 4; ++et)
#pragma unroll
    for (int r = 0; r < 8; ++r) {
      float v = O[et][r] / lsum[r];
      int qrow = qt * 16 + r + 8 * lg;
      AO[(size_t)(b * N_ + qrow) * (H_ * DO_) + h * DO_ + et * 16 + ln] =
          (__bf16)v;
    }
}

// ------------------------- final linear + bias + ELU -----------------------
// grid (B*N/16, DLIN/16), block 32
__global__ __launch_bounds__(32)
void linear_kernel(const __bf16* __restrict__ AO, const __bf16* __restrict__ Wlt,
                   const float* __restrict__ blin, float* __restrict__ out) {
  const int rt = blockIdx.x;            // 0..511
  const int nt = blockIdx.y;            // 0..7
  const int l  = threadIdx.x & 31;
  const int ln = l & 15, lg = l >> 4;
  const int KD = H_ * DO_;              // 512

  float bb = blin[nt * 16 + ln];
  v8f c;
#pragma unroll
  for (int r = 0; r < 8; ++r) c[r] = bb;

#pragma unroll
  for (int k0 = 0; k0 < KD; k0 += 32) {
    v16bf a = load_frag(AO + (size_t)(rt * 16) * KD + k0, KD);
    v16bf w = load_frag(Wlt + (size_t)(nt * 16) * KD + k0, KD);
    c = wmma_bf(a, w, c);
  }
#pragma unroll
  for (int r = 0; r < 8; ++r) {
    float v = c[r];
    v = v > 0.f ? v : expm1f(v);        // ELU (alpha = 1)
    out[(size_t)(rt * 16 + r + 8 * lg) * DLIN_ + nt * 16 + ln] = v;
  }
}

// ---------------------------------------------------------------------------

extern "C" void kernel_launch(void* const* d_in, const int* in_sizes, int n_in,
                              void* d_out, int out_size, void* d_ws, size_t ws_size,
                              hipStream_t stream) {
  (void)in_sizes; (void)n_in; (void)out_size; (void)ws_size;
  const float* X     = (const float*)d_in[0];
  const int*   masks = (const int*)d_in[1];
  const int*   adj   = (const int*)d_in[2];
  const float* Wq    = (const float*)d_in[3];
  const float* Wk    = (const float*)d_in[4];
  const float* Wv    = (const float*)d_in[5];
  const float* bq    = (const float*)d_in[6];
  const float* bk    = (const float*)d_in[7];
  const float* bv    = (const float*)d_in[8];
  const float* Wlin  = (const float*)d_in[9];
  const float* blin  = (const float*)d_in[10];
  float* out = (float*)d_out;

  char* ws = (char*)d_ws;
  size_t off = 0;
  auto alloc = [&](size_t bytes) -> void* {
    void* p = ws + off;
    off = (off + bytes + 255) & ~(size_t)255;
    return p;
  };

  const int nX   = B_ * N_ * DIN_;          // 1,048,576
  const int nW   = H_ * DIN_ * DH_;         // 65,536 (each of Wq/Wk/Wv)
  const int nWl  = (H_ * DO_) * DLIN_;      // 65,536
  const int nQK  = B_ * H_ * N_ * DH_;      // 4,194,304
  const int nAO  = B_ * N_ * H_ * DO_;      // 4,194,304

  __bf16* Xh  = (__bf16*)alloc((size_t)nX  * 2);
  __bf16* Wqt = (__bf16*)alloc((size_t)nW  * 2);
  __bf16* Wkt = (__bf16*)alloc((size_t)nW  * 2);
  __bf16* Wvt = (__bf16*)alloc((size_t)nW  * 2);
  __bf16* Wlt = (__bf16*)alloc((size_t)nWl * 2);
  __bf16* Qb  = (__bf16*)alloc((size_t)nQK * 2);
  __bf16* Kb  = (__bf16*)alloc((size_t)nQK * 2);
  __bf16* Vt  = (__bf16*)alloc((size_t)nQK * 2);
  __bf16* AO  = (__bf16*)alloc((size_t)nAO * 2);

  // 1) cast node features, transpose + cast all weights to bf16
  cast_bf16_kernel<<<(nX + 255) / 256, 256, 0, stream>>>(X, Xh, nX);
  transpose_bf16_kernel<<<(nW + 255) / 256, 256, 0, stream>>>(Wq, Wqt, H_, DIN_, DH_);
  transpose_bf16_kernel<<<(nW + 255) / 256, 256, 0, stream>>>(Wk, Wkt, H_, DIN_, DH_);
  transpose_bf16_kernel<<<(nW + 255) / 256, 256, 0, stream>>>(Wv, Wvt, H_, DIN_, DO_);
  transpose_bf16_kernel<<<(nWl + 255) / 256, 256, 0, stream>>>(Wlin, Wlt, 1, H_ * DO_, DLIN_);

  // 2) Q/K/V projections (WMMA)
  qkv_kernel<<<dim3(N_ / 16, B_ * H_, 3), 128, 0, stream>>>(
      Xh, Wqt, Wkt, Wvt, bq, bk, bv, Qb, Kb, Vt);

  // 3) fused masked attention (WMMA + online softmax + TDM staging)
  attn_kernel<<<dim3(N_ / 16, B_ * H_), 32, 0, stream>>>(
      Qb, Kb, Vt, adj, masks, AO);

  // 4) final linear + bias + ELU (WMMA)
  linear_kernel<<<dim3((B_ * N_) / 16, DLIN_ / 16), 32, 0, stream>>>(
      AO, Wlt, blin, out);
}
